// VanillaRNN_77850577207535
// MI455X (gfx1250) — compile-verified, matrix operand order checked by reading, not run on hardware
//
#include <hip/hip_runtime.h>

#define B_ 1024
#define T_ 1024
#define H_ 512
#define C_ 10

#define BM 32                    // batch rows per workgroup
#define NWAVES 16
#define THREADS (NWAVES * 32)

#define KPAD 8                   // 8 bf16 = 16B pad per row -> pitch 1040B (== 16 mod 256: conflict-free b128 tile loads)
#define ROWP (H_ + KPAD)         // 520 bf16 elements per row
#define W_SLOTS 256              // 256 of 512 W columns cached in LDS (the Nsub=0 tile of every wave)

typedef __attribute__((ext_vector_type(16))) __bf16 v16bf;
typedef __attribute__((ext_vector_type(8)))  float  v8f;
typedef __attribute__((ext_vector_type(4)))  unsigned int u32x4;

// 32-byte tile load as two 16B chunks (pitch is only 16B aligned)
__device__ __forceinline__ v16bf ld_tile(const __bf16* p) {
  union { u32x4 q[2]; v16bf v; } u;
  const u32x4* s = (const u32x4*)p;
  u.q[0] = s[0];
  u.q[1] = s[1];
  return u.v;
}

// tanh(v) = 1 - 2/(exp2(v*2*log2 e)+1); saturates correctly at +-1
__device__ __forceinline__ float fast_tanh(float v) {
  float e = __builtin_amdgcn_exp2f(v * 2.8853900817779268f);
  return 1.0f - 2.0f * __builtin_amdgcn_rcpf(e + 1.0f);
}

// W_hh [k][j] f32 (row-major) -> Wt [j][k] bf16 (column-major of W_hh, contiguous in k)
__global__ __launch_bounds__(256) void prep_w(const float* __restrict__ Whh,
                                              __bf16* __restrict__ Wt) {
  int idx = blockIdx.x * 256 + threadIdx.x;
  if (idx < H_ * H_) {
    int j = idx / H_, k = idx % H_;
    Wt[idx] = (__bf16)Whh[k * H_ + j];
  }
}

__global__ __launch_bounds__(THREADS)
void rnn_kernel(const float* __restrict__ x,      // [B,T]
                const float* __restrict__ Whx,    // [H]
                const float* __restrict__ bh,     // [H]
                const __bf16* __restrict__ Wt,    // [H cols][H k] bf16 (transposed W_hh)
                const float* __restrict__ Wph,    // [H,C]
                const float* __restrict__ bp,     // [C]
                float* __restrict__ out)          // [B,C]
{
  __shared__ __bf16 w_lds[W_SLOTS * ROWP];   // 266,240 B: columns j with (j%32)<16
  __shared__ __bf16 h_lds[BM * ROWP];        //  33,280 B: h tile, bf16, row-major in k
  __shared__ float  x_s[2][BM];              // double-buffered x_t for this batch tile

  const int tid  = threadIdx.x;
  const int wave = tid >> 5;
  const int lane = tid & 31;
  const int b0   = blockIdx.x * BM;

  // ---- fill LDS W cache: slot s <-> column j = (s/16)*32 + (s%16); 2 threads per column
  {
    int s    = tid >> 1;
    int half = tid & 1;
    int j    = ((s >> 4) << 5) + (s & 15);
    const u32x4* src = (const u32x4*)(Wt + j * H_ + half * 256);
    u32x4*       dst = (u32x4*)(w_lds + s * ROWP + half * 256);
    #pragma unroll
    for (int i = 0; i < 32; ++i) dst[i] = src[i];   // 512B per thread
  }
  // ---- zero h0
  for (int i = tid; i < (BM * ROWP) / 2; i += THREADS) ((unsigned int*)h_lds)[i] = 0u;
  // ---- x for t=0
  if (tid < BM) x_s[0][tid] = x[(b0 + tid) * T_];
  __syncthreads();

  // per-wave / per-lane constants
  const int n0    = wave * 32;
  const int nA    = n0 + (lane & 15);        // Nsub=0 column (LDS-cached W)
  const int nB    = n0 + 16 + (lane & 15);   // Nsub=1 column (streamed from L2)
  const int khalf = (lane >> 4) << 4;        // lanes 16-31 hold K+16..K+31

  const float whx0 = Whx[nA], whx1 = Whx[nB];
  const float bh0  = bh[nA],  bh1  = bh[nB];

  const __bf16* als0 = h_lds + (lane & 15) * ROWP + khalf;            // A tile, M=0..15
  const __bf16* als1 = als0 + 16 * ROWP;                              // A tile, M=16..31
  const __bf16* wls  = w_lds + (wave * 16 + (lane & 15)) * ROWP + khalf;  // B tile (LDS)
  const __bf16* wgl  = Wt + nB * H_ + khalf;                          // B tile (L2 stream)

  for (int t = 0; t < T_; ++t) {
    if (tid < BM && t + 1 < T_)
      x_s[(t + 1) & 1][tid] = x[(b0 + tid) * T_ + t + 1];   // prefetch next x_t

    v8f acc00 = {}, acc01 = {}, acc10 = {}, acc11 = {};
    #pragma unroll 4
    for (int kk = 0; kk < H_; kk += 32) {
      v16bf a0 = ld_tile(als0 + kk);
      v16bf a1 = ld_tile(als1 + kk);
      v16bf bL = ld_tile(wls + kk);
      v16bf bG = ld_tile(wgl + kk);
      acc00 = __builtin_amdgcn_wmma_f32_16x16x32_bf16(false, a0, false, bL, (short)0, acc00, false, false);
      acc10 = __builtin_amdgcn_wmma_f32_16x16x32_bf16(false, a1, false, bL, (short)0, acc10, false, false);
      acc01 = __builtin_amdgcn_wmma_f32_16x16x32_bf16(false, a0, false, bG, (short)0, acc01, false, false);
      acc11 = __builtin_amdgcn_wmma_f32_16x16x32_bf16(false, a1, false, bG, (short)0, acc11, false, false);
    }
    __syncthreads();   // all reads of h_lds done

    // h_new = tanh(acc + x_t[m]*W_hx[n] + b_h[n]); write back as bf16
    const float* xs   = x_s[t & 1];
    const int    mofs = (lane >> 4) << 3;   // C/D layout: lanes 16-31 hold M = r+8
    #pragma unroll
    for (int r = 0; r < 8; ++r) {
      int m0 = r + mofs, m1 = m0 + 16;
      float xm0 = xs[m0], xm1 = xs[m1];
      h_lds[m0 * ROWP + nA] = (__bf16)fast_tanh(acc00[r] + xm0 * whx0 + bh0);
      h_lds[m0 * ROWP + nB] = (__bf16)fast_tanh(acc01[r] + xm0 * whx1 + bh1);
      h_lds[m1 * ROWP + nA] = (__bf16)fast_tanh(acc10[r] + xm1 * whx0 + bh0);
      h_lds[m1 * ROWP + nB] = (__bf16)fast_tanh(acc11[r] + xm1 * whx1 + bh1);
    }
    __syncthreads();
  }

  // ---- final projection: out[b0+m][c] = h_T[m,:] . W_ph[:,c] + b_p[c]  (one-shot, tiny)
  if (tid < BM * C_) {
    int m = tid / C_, c = tid % C_;
    float accv = bp[c];
    const __bf16* hr = h_lds + m * ROWP;
    #pragma unroll 8
    for (int j = 0; j < H_; ++j)
      accv += (float)hr[j] * Wph[j * C_ + c];
    out[(b0 + m) * C_ + c] = accv;
  }
}

extern "C" void kernel_launch(void* const* d_in, const int* in_sizes, int n_in,
                              void* d_out, int out_size, void* d_ws, size_t ws_size,
                              hipStream_t stream) {
  const float* x   = (const float*)d_in[0];   // [B,T]
  const float* Whx = (const float*)d_in[1];   // [1,H]
  const float* Whh = (const float*)d_in[2];   // [H,H]
  const float* bh  = (const float*)d_in[3];   // [H]
  const float* Wph = (const float*)d_in[4];   // [H,C]
  const float* bp  = (const float*)d_in[5];   // [C]
  float* out = (float*)d_out;

  __bf16* Wt = (__bf16*)d_ws;                 // H*H bf16 = 512 KB scratch

  prep_w<<<(H_ * H_ + 255) / 256, 256, 0, stream>>>(Whh, Wt);
  rnn_kernel<<<B_ / BM, THREADS, 0, stream>>>(x, Whx, bh, Wt, Wph, bp, out);
}
